// MultiheadAttention_78924319031414
// MI455X (gfx1250) — matrix operand order repaired and checked by reading, not compiled, not run on hardware
//
#include <hip/hip_runtime.h>

typedef _Float16 f16;
typedef __attribute__((ext_vector_type(16))) _Float16 v16h;
typedef __attribute__((ext_vector_type(8)))  float    v8f;

#define BATCH   4
#define HEADS   16
#define S_LEN   2048
#define DK      64
#define DMODEL  1024

// ---- CDNA5 async global->LDS copy (ASYNCcnt-tracked), with safe fallback ----
#if defined(__gfx1250__) && __has_builtin(__builtin_amdgcn_global_load_async_to_lds_b128)
#define USE_ASYNC_LDS 1

// Declare the LLVM intrinsic directly (asm-label trick) so we control the
// pointer parameter types: they lower to ptr addrspace(1) / ptr addrspace(3).
typedef int v4i_a __attribute__((ext_vector_type(4)));
typedef const __attribute__((address_space(1))) v4i_a* gas_v4i;
typedef __attribute__((address_space(3)))       v4i_a* las_v4i;
__device__ void llvm_amdgcn_global_load_async_to_lds_b128(gas_v4i, las_v4i, int, int)
    __asm("llvm.amdgcn.global.load.async.to.lds.b128");

__device__ __forceinline__ void async_cp16(const void* g, void* l) {
    // flat global address == AS1 address; low 32 bits of a flat LDS
    // address are the LDS byte offset (aperture rules).
    llvm_amdgcn_global_load_async_to_lds_b128(
        (gas_v4i)(unsigned long long)g,
        (las_v4i)(unsigned int)(unsigned long long)l,
        0, 0);
}
__device__ __forceinline__ void async_wait0() {
#if __has_builtin(__builtin_amdgcn_s_wait_asynccnt)
    __builtin_amdgcn_s_wait_asynccnt(0);
#else
    asm volatile("s_wait_asynccnt 0x0" ::: "memory");
#endif
}
#else
#define USE_ASYNC_LDS 0
#endif

// ---------------------------------------------------------------------------
// Tiled WMMA GEMM:  Y[M=8192, N=1024] = X[M,K=1024] @ W[K,N]
// Block: 256 threads (8 waves), tile 128(M) x 64(N), K staged in 32-chunks
// through LDS (convert to f16 if needed).  Each wave: 16x64, 4 accumulators.
// HEADSPLIT: write f16 into [B,H,S,DK] layout; else f32 into plain [M,N].
// ---------------------------------------------------------------------------
template <typename TX, bool HEADSPLIT>
__global__ __launch_bounds__(256) void gemm_wmma_kernel(
    const TX* __restrict__ X, const float* __restrict__ W,
    void* __restrict__ Out, float scale)
{
    __shared__ f16 As[128 * 32];
    __shared__ f16 Bs[32 * 64];

    const int tid  = threadIdx.x;
    const int lane = tid & 31;
    const int wave = tid >> 5;
    const int half = lane >> 4;
    const int nn   = lane & 15;

    const int m0 = blockIdx.x * 128;
    const int n0 = blockIdx.y * 64;

    v8f acc[4];
#pragma unroll
    for (int dt = 0; dt < 4; ++dt)
#pragma unroll
        for (int r = 0; r < 8; ++r) acc[dt][r] = 0.f;

    for (int k0 = 0; k0 < DMODEL; k0 += 32) {
        // ---- stage A tile: 128 rows x 32 k ----
        {
            const int row = tid >> 1;
            const int kb  = (tid & 1) * 16;
            const TX* src = X + (size_t)(m0 + row) * DMODEL + k0 + kb;
#if USE_ASYNC_LDS
            if constexpr (sizeof(TX) == 2) {
                // already f16: raw async byte copy, no VGPR round-trip
                f16* dst = &As[row * 32 + kb];
                async_cp16((const void*)src,       (void*)dst);
                async_cp16((const void*)(src + 8), (void*)(dst + 8));
            } else
#endif
            {
#pragma unroll
                for (int j = 0; j < 16; ++j)
                    As[row * 32 + kb + j] = (f16)src[j];
            }
            if (k0 + 32 < DMODEL)
                __builtin_prefetch((const void*)(src + 32), 0, 1); // global_prefetch_b8
        }
        // ---- stage B tile: 32 k x 64 n (convert f32 -> f16) ----
        {
            const int kr = tid >> 3;
            const int cb = (tid & 7) * 8;
            const float* src = W + (size_t)(k0 + kr) * DMODEL + n0 + cb;
#pragma unroll
            for (int j = 0; j < 8; ++j)
                Bs[kr * 64 + cb + j] = (f16)src[j];
        }
#if USE_ASYNC_LDS
        if constexpr (sizeof(TX) == 2) async_wait0();
#endif
        __syncthreads();

        // A fragment: 16(M) x 32(K) f16.  lane<16: K 0..15; lane>=16: K 16..31
        v16h a;
#pragma unroll
        for (int i = 0; i < 8; ++i) {
            const int kk = 2 * i + 16 * half;
            a[2 * i + 0] = As[(wave * 16 + nn) * 32 + kk + 0];
            a[2 * i + 1] = As[(wave * 16 + nn) * 32 + kk + 1];
        }
#pragma unroll
        for (int dt = 0; dt < 4; ++dt) {
            // B fragment: 32(K) x 16(N) f16. lane gives N; K = 2i + 16*half
            v16h bf;
#pragma unroll
            for (int i = 0; i < 8; ++i) {
                const int kk = 2 * i + 16 * half;
                bf[2 * i + 0] = Bs[(kk + 0) * 64 + dt * 16 + nn];
                bf[2 * i + 1] = Bs[(kk + 1) * 64 + dt * 16 + nn];
            }
            acc[dt] = __builtin_amdgcn_wmma_f32_16x16x32_f16(
                false, a, false, bf, (short)0, acc[dt], false, false);
        }
        __syncthreads();
    }

    // ---- store (C layout: lane = N, VGPR r -> M = r + 8*half) ----
#pragma unroll
    for (int dt = 0; dt < 4; ++dt) {
#pragma unroll
        for (int r = 0; r < 8; ++r) {
            const int m   = r + 8 * half;
            const int row = m0 + wave * 16 + m;
            const int col = n0 + dt * 16 + nn;
            const float val = acc[dt][r] * scale;
            if (HEADSPLIT) {
                const int b = row >> 11, s = row & (S_LEN - 1);
                const int h = col >> 6,  d = col & (DK - 1);
                ((f16*)Out)[((size_t)(b * HEADS + h) * S_LEN + s) * DK + d] = (f16)val;
            } else {
                ((float*)Out)[(size_t)row * DMODEL + col] = val;
            }
        }
    }
}

// ---------------------------------------------------------------------------
// Causal attention per (b, h, 16-query tile): one wave.
// pass1: streaming row-max + sum-exp over QK^T tiles (WMMA + shfl reductions)
// pass2: recompute scores, write normalized probs, P relayout via LDS, P@V.
// V tiles are staged through LDS with async global->LDS copies (CDNA5 path),
// overlapped with the score WMMAs of the same iteration.
// ---------------------------------------------------------------------------
__global__ __launch_bounds__(32) void attn_wmma_kernel(
    const f16* __restrict__ Qh, const f16* __restrict__ Kh,
    const f16* __restrict__ Vh, float* __restrict__ attn,
    f16* __restrict__ Oc)
{
    __shared__ f16 Pt[16 * 32];
#if USE_ASYNC_LDS
    __shared__ f16 Vt[32 * DK];
#endif

    const int lane = threadIdx.x;
    const int half = lane >> 4;
    const int nn   = lane & 15;

    const int qt = blockIdx.x;
    const int h  = blockIdx.y;
    const int b  = blockIdx.z;
    const int bh = b * HEADS + h;
    const int q0 = qt * 16;

    const size_t kvBase = (size_t)bh * S_LEN * DK;
    float* attnRow = attn + (size_t)bh * S_LEN * S_LEN;

    // ---- zero-fill the causally-masked columns [ (qt+1)*16, S ) ----
    {
        const int cstart = (qt + 1) * 16;
        const float4 z = make_float4(0.f, 0.f, 0.f, 0.f);
        for (int r = 0; r < 16; ++r) {
            float* rowp = attnRow + (size_t)(q0 + r) * S_LEN;
            for (int c = cstart + lane * 4; c < S_LEN; c += 128)
                *reinterpret_cast<float4*>(rowp + c) = z;
        }
    }

    // ---- Q fragments (2 chunks of K=32 over DK=64); 1/T pre-folded in Qh ----
    v16h qa[2];
#pragma unroll
    for (int c = 0; c < 2; ++c) {
#pragma unroll
        for (int i = 0; i < 8; ++i) {
            const int kd = 2 * i + 16 * half + 32 * c;
            const f16* p = Qh + kvBase + (size_t)(q0 + nn) * DK + kd;
            qa[c][2 * i + 0] = p[0];
            qa[c][2 * i + 1] = p[1];
        }
    }

    float mrow[8], lrow[8];
#pragma unroll
    for (int r = 0; r < 8; ++r) { mrow[r] = -1e30f; lrow[r] = 0.f; }

    // ---------------- pass 1: online max / sum-exp ----------------
    for (int kt = 0; kt <= qt; ++kt) {
        v8f s;
#pragma unroll
        for (int r = 0; r < 8; ++r) s[r] = 0.f;
#pragma unroll
        for (int c = 0; c < 2; ++c) {
            v16h kb;
#pragma unroll
            for (int i = 0; i < 8; ++i) {
                const int kd = 2 * i + 16 * half + 32 * c;
                const f16* p = Kh + kvBase + (size_t)(kt * 16 + nn) * DK + kd;
                kb[2 * i + 0] = p[0];
                kb[2 * i + 1] = p[1];
            }
            s = __builtin_amdgcn_wmma_f32_16x16x32_f16(
                false, qa[c], false, kb, (short)0, s, false, false);
        }
        const int key = kt * 16 + nn;
#pragma unroll
        for (int r = 0; r < 8; ++r) {
            const int q = q0 + r + 8 * half;
            const float sv = (key > q) ? -1e30f : s[r];
            float tm = sv;                               // row max across 16 lanes
            tm = fmaxf(tm, __shfl_xor(tm, 1, 32));
            tm = fmaxf(tm, __shfl_xor(tm, 2, 32));
            tm = fmaxf(tm, __shfl_xor(tm, 4, 32));
            tm = fmaxf(tm, __shfl_xor(tm, 8, 32));
            const float nm = fmaxf(mrow[r], tm);
            float e = __expf(sv - nm);
            e += __shfl_xor(e, 1, 32);
            e += __shfl_xor(e, 2, 32);
            e += __shfl_xor(e, 4, 32);
            e += __shfl_xor(e, 8, 32);
            lrow[r] = lrow[r] * __expf(mrow[r] - nm) + e;
            mrow[r] = nm;
        }
    }

    float linv[8];
#pragma unroll
    for (int r = 0; r < 8; ++r) linv[r] = 1.0f / lrow[r];

    // ---------------- pass 2: probs + P@V ----------------
    v8f oacc[4];
#pragma unroll
    for (int dt = 0; dt < 4; ++dt)
#pragma unroll
        for (int r = 0; r < 8; ++r) oacc[dt][r] = 0.f;

    for (int ktp = 0; ktp <= (qt >> 1); ++ktp) {
#if USE_ASYNC_LDS
        // async-stage the 32x64 V tile (lane -> one 128B key row),
        // overlapped with the score WMMAs / softmax below
        {
            const f16* src = Vh + kvBase + (size_t)(ktp * 32 + lane) * DK;
            f16* dst = &Vt[lane * DK];
#pragma unroll
            for (int j = 0; j < 8; ++j)
                async_cp16((const void*)(src + j * 8), (void*)(dst + j * 8));
        }
#endif
#pragma unroll
        for (int sub = 0; sub < 2; ++sub) {
            const int kt = 2 * ktp + sub;
            if (kt <= qt) {
                v8f s;
#pragma unroll
                for (int r = 0; r < 8; ++r) s[r] = 0.f;
#pragma unroll
                for (int c = 0; c < 2; ++c) {
                    v16h kb;
#pragma unroll
                    for (int i = 0; i < 8; ++i) {
                        const int kd = 2 * i + 16 * half + 32 * c;
                        const f16* p = Kh + kvBase + (size_t)(kt * 16 + nn) * DK + kd;
                        kb[2 * i + 0] = p[0];
                        kb[2 * i + 1] = p[1];
                    }
                    s = __builtin_amdgcn_wmma_f32_16x16x32_f16(
                        false, qa[c], false, kb, (short)0, s, false, false);
                }
                const int key = kt * 16 + nn;
#pragma unroll
                for (int r = 0; r < 8; ++r) {
                    const int m = r + 8 * half;
                    const int q = q0 + m;
                    const float pv = (key > q) ? 0.f
                                               : __expf(s[r] - mrow[r]) * linv[r];
                    attnRow[(size_t)q * S_LEN + key] = pv;
                    Pt[m * 32 + sub * 16 + nn] = (f16)pv;
                }
            } else {
#pragma unroll
                for (int r = 0; r < 8; ++r) {
                    const int m = r + 8 * half;
                    Pt[m * 32 + sub * 16 + nn] = (f16)0.f;
                }
            }
        }
        __syncthreads();

        // P as A-fragment (16x32) from LDS
        v16h pa;
#pragma unroll
        for (int i = 0; i < 8; ++i) {
            const int kk = 2 * i + 16 * half;
            pa[2 * i + 0] = Pt[nn * 32 + kk + 0];
            pa[2 * i + 1] = Pt[nn * 32 + kk + 1];
        }
#if USE_ASYNC_LDS
        async_wait0();   // V tile resident in LDS
#endif
        // V as B-fragments (32 keys x 16 d), 4 d-tiles
#pragma unroll
        for (int dt = 0; dt < 4; ++dt) {
            v16h vb;
#pragma unroll
            for (int i = 0; i < 8; ++i) {
                const int kk = 2 * i + 16 * half;
#if USE_ASYNC_LDS
                vb[2 * i + 0] = Vt[(kk + 0) * DK + dt * 16 + nn];
                vb[2 * i + 1] = Vt[(kk + 1) * DK + dt * 16 + nn];
#else
                const int key = ktp * 32 + kk;
                const f16* p = Vh + kvBase + (size_t)key * DK + dt * 16 + nn;
                vb[2 * i + 0] = p[0];
                vb[2 * i + 1] = p[DK];
#endif
            }
            oacc[dt] = __builtin_amdgcn_wmma_f32_16x16x32_f16(
                false, pa, false, vb, (short)0, oacc[dt], false, false);
        }
        __syncthreads();  // protects Pt/Vt WAR for the next iteration
    }

    // ---- store O in concat-head layout [B*S, H*DK] (f16) ----
#pragma unroll
    for (int dt = 0; dt < 4; ++dt) {
#pragma unroll
        for (int r = 0; r < 8; ++r) {
            const int m   = r + 8 * half;
            const int row = b * S_LEN + q0 + m;
            const int col = h * DK + dt * 16 + nn;
            Oc[(size_t)row * DMODEL + col] = (f16)oacc[dt][r];
        }
    }
}

// ---------------------------------------------------------------------------
extern "C" void kernel_launch(void* const* d_in, const int* in_sizes, int n_in,
                              void* d_out, int out_size, void* d_ws, size_t ws_size,
                              hipStream_t stream)
{
    (void)in_sizes; (void)n_in; (void)out_size; (void)ws_size;

    const float* q  = (const float*)d_in[0];
    const float* k  = (const float*)d_in[1];
    const float* v  = (const float*)d_in[2];
    /* d_in[3] = causal mask (computed analytically in-kernel) */
    const float* Wq = (const float*)d_in[4];
    const float* Wk = (const float*)d_in[5];
    const float* Wv = (const float*)d_in[6];
    const float* Wo = (const float*)d_in[7];

    const size_t headElems = (size_t)BATCH * HEADS * S_LEN * DK; // 8,388,608
    f16* Qh = (f16*)d_ws;
    f16* Kh = Qh + headElems;
    f16* Vh = Kh + headElems;
    f16* Oc = Vh + headElems;

    float* out  = (float*)d_out;                                  // [B*S, DMODEL]
    float* attn = out + (size_t)BATCH * S_LEN * DMODEL;           // [B,H,S,S]

    const dim3 ggrid(64, 16, 1);       // M=8192/128, N=1024/64
    const float invT = 1.0f / 32.0f;   // 1/sqrt(H*DK) folded into Qh

    gemm_wmma_kernel<float, true ><<<ggrid, 256, 0, stream>>>(q, Wq, (void*)Qh, invT);
    gemm_wmma_kernel<float, true ><<<ggrid, 256, 0, stream>>>(k, Wk, (void*)Kh, 1.0f);
    gemm_wmma_kernel<float, true ><<<ggrid, 256, 0, stream>>>(v, Wv, (void*)Vh, 1.0f);

    attn_wmma_kernel<<<dim3(S_LEN / 16, HEADS, BATCH), 32, 0, stream>>>(
        Qh, Kh, Vh, attn, Oc);

    gemm_wmma_kernel<_Float16, false><<<ggrid, 256, 0, stream>>>(Oc, Wo, (void*)out, 1.0f);
}
